// SelfAttention_21947282882679
// MI455X (gfx1250) — matrix operand order repaired and checked by reading, not compiled, not run on hardware
//
#include <hip/hip_runtime.h>
#include <hip/hip_bf16.h>

// ---------------- problem constants ----------------
#define BB   2
#define SQ   2048
#define HH   4096
#define NH   32
#define NG   2
#define HN   128
#define REP  (NH / NG)          // 16
#define QD   (NH * HN)          // 4096
#define KD   (NG * HN)          // 256
#define QKV  (QD + 2 * KD)      // 4608
#define TOK  (BB * SQ)          // 4096
#define QSCALE 0.08838834764831845f   // 1/sqrt(128); folds /sqrt(HN) and *LAYER_NUMBER

typedef __attribute__((ext_vector_type(16))) _Float16 v16h;
typedef __attribute__((ext_vector_type(8)))  _Float16 v8h;
typedef __attribute__((ext_vector_type(4)))  _Float16 v4h;
typedef __attribute__((ext_vector_type(8)))  float    v8f;
typedef __attribute__((ext_vector_type(4)))  int      v4i;

union V16 { v16h v; v8h h[2]; };

// CDNA5 async global->LDS path (ASYNCcnt-tracked), with safe fallback.
#if defined(__has_builtin)
#if __has_builtin(__builtin_amdgcn_global_load_async_to_lds_b128) && \
    __has_builtin(__builtin_amdgcn_s_wait_asynccnt)
#define HAVE_ASYNC_LDS 1
#endif
#endif
#ifndef HAVE_ASYNC_LDS
#define HAVE_ASYNC_LDS 0
#endif

#if HAVE_ASYNC_LDS
typedef __attribute__((address_space(1))) v4i gas_v4i;   // global int4
typedef __attribute__((address_space(3))) v4i las_v4i;   // LDS int4
#endif

__device__ __forceinline__ void async_copy16(const _Float16* gsrc, _Float16* ldst) {
#if HAVE_ASYNC_LDS
  __builtin_amdgcn_global_load_async_to_lds_b128(
      (gas_v4i*)(gsrc), (las_v4i*)(ldst), 0, 0);
#else
  *(v8h*)ldst = *(const v8h*)gsrc;
#endif
}

__device__ __forceinline__ void async_drain() {
#if HAVE_ASYNC_LDS
  __builtin_amdgcn_s_wait_asynccnt(0);
#endif
}

__device__ inline v8f wmma_f16(v16h a, v16h b, v8f c) {
  // D = A(16x32 f16) * B(32x16 f16) + C(16x16 f32)
  return __builtin_amdgcn_wmma_f32_16x16x32_f16(
      /*neg_a=*/false, a, /*neg_b=*/false, b,
      /*c_mod=*/(short)0, c, /*reuse_a=*/false, /*reuse_b=*/false);
}

// ---------------- fp32 -> f16 cast ----------------
__global__ void cast_f32_f16_k(const float* __restrict__ in,
                               _Float16* __restrict__ out, long n) {
  long i = ((long)blockIdx.x * blockDim.x + threadIdx.x) * 4;
  long stride = (long)gridDim.x * blockDim.x * 4;
  for (; i < n; i += stride) {
    float4 v = *(const float4*)(in + i);
    v4h o;
    o[0] = (_Float16)v.x; o[1] = (_Float16)v.y;
    o[2] = (_Float16)v.z; o[3] = (_Float16)v.w;
    *(v4h*)(out + i) = o;
  }
}

// ---------------- generic WMMA GEMM: C[M,N] = A[M,K] * W[N,K]^T (+bias) ----------------
// Block: 256 threads = 8 waves arranged 2(M) x 4(N); wave computes 64x64 via 4x4 WMMA
// tiles -> 16 WMMAs per 8 fragment loads (2x the MAC/byte of a 32x32 wave tile).
__global__ __launch_bounds__(256)
void gemm_wmma_k(const _Float16* __restrict__ A, const _Float16* __restrict__ W,
                 const float* __restrict__ bias, float* __restrict__ outf,
                 _Float16* __restrict__ outh, int M, int N, int K) {
  const int lane = threadIdx.x & 31;
  const int w    = threadIdx.x >> 5;
  const int wm   = w & 1, wn = w >> 1;
  const int Mb   = blockIdx.y * 128 + wm * 64;
  const int Nb   = blockIdx.x * 256 + wn * 64;
  const int col  = lane & 15;
  const int hi   = lane >> 4;        // 0: lanes 0-15, 1: lanes 16-31
  const int kb8  = hi * 8;           // A-frag K sub-offset (halves)
  const int kb16 = hi * 16;          // unused; B-frag uses kb8 like A

  v8f acc[4][4];
#pragma unroll
  for (int i = 0; i < 4; ++i)
#pragma unroll
    for (int j = 0; j < 4; ++j) acc[i][j] = (v8f){};

  const _Float16* arow[4];
  const _Float16* wrow[4];
#pragma unroll
  for (int i = 0; i < 4; ++i) arow[i] = A + (long)(Mb + i * 16 + col) * K;
#pragma unroll
  for (int j = 0; j < 4; ++j) wrow[j] = W + (long)(Nb + j * 16 + col) * K;

  for (int k0 = 0; k0 < K; k0 += 32) {
    V16 a[4];
    v16h b[4];
#pragma unroll
    for (int i = 0; i < 4; ++i) {
      a[i].h[0] = *(const v8h*)(arow[i] + k0 + kb8);
      a[i].h[1] = *(const v8h*)(arow[i] + k0 + 16 + kb8);
    }
#pragma unroll
    for (int j = 0; j < 4; ++j) { V16 t; t.h[0] = *(const v8h*)(wrow[j] + k0 + kb8); t.h[1] = *(const v8h*)(wrow[j] + k0 + 16 + kb8); b[j] = t.v; }
#pragma unroll
    for (int i = 0; i < 4; ++i)
#pragma unroll
      for (int j = 0; j < 4; ++j)
        acc[i][j] = wmma_f16(a[i].v, b[j], acc[i][j]);
    __builtin_prefetch(arow[0] + k0 + 256, 0, 3);
    __builtin_prefetch(wrow[0] + k0 + 256, 0, 3);
  }

#pragma unroll
  for (int j = 0; j < 4; ++j) {
    const int nbase = Nb + j * 16 + col;
    const float bv = bias ? bias[nbase] : 0.0f;
#pragma unroll
    for (int i = 0; i < 4; ++i) {
#pragma unroll
      for (int r = 0; r < 8; ++r) {
        const int m = Mb + i * 16 + r + hi * 8;
        const long off = (long)m * N + nbase;
        const float val = acc[i][j][r] + bv;
        if (outf) outf[off] = val;
        else      outh[off] = (_Float16)val;
      }
    }
  }
}

// ---------------- RoPE + scatter q/k ----------------
__global__ void rope_qk_k(const _Float16* __restrict__ mixed,
                          const float* __restrict__ cache,
                          _Float16* __restrict__ q, _Float16* __restrict__ k) {
  long idx = (long)blockIdx.x * blockDim.x + threadIdx.x;
  const long total = (long)TOK * (NH + NG) * 64;
  if (idx >= total) return;
  const int  pair = idx & 63;
  const int  hidx = (int)((idx >> 6) % (NH + NG));
  const long tok  = idx / (64 * (NH + NG));
  const int  b_i  = (int)(tok / SQ);
  const int  s    = (int)(tok % SQ);
  const int  d0   = pair * 2, d1 = d0 + 1;

  const _Float16* src;
  _Float16* dst;
  float scale;
  if (hidx < NH) {
    src = mixed + tok * QKV + (long)hidx * HN;
    dst = q + (((long)b_i * NH + hidx) * SQ + s) * HN;
    scale = QSCALE;
  } else {
    const int g = hidx - NH;
    src = mixed + tok * QKV + QD + (long)g * HN;
    dst = k + (((long)b_i * NG + g) * SQ + s) * HN;
    scale = 1.0f;
  }
  const float x0 = (float)src[d0], x1 = (float)src[d1];
  if (pair < 32) {
    const float c  = cache[((long)s * 32 + pair) * 2 + 0];
    const float sn = cache[((long)s * 32 + pair) * 2 + 1];
    dst[d0] = (_Float16)((x0 * c - x1 * sn) * scale);
    dst[d1] = (_Float16)((x1 * c + x0 * sn) * scale);
  } else {
    dst[d0] = (_Float16)(x0 * scale);
    dst[d1] = (_Float16)(x1 * scale);
  }
}

// ---------------- V transpose: vT[b][g][hn][sq] ----------------
__global__ void v_trans_k(const _Float16* __restrict__ mixed,
                          _Float16* __restrict__ vT) {
  long idx = (long)blockIdx.x * blockDim.x + threadIdx.x;
  const long total = (long)TOK * NG * HN;
  if (idx >= total) return;
  const int  d   = (int)(idx % HN);
  const int  g   = (int)((idx / HN) % NG);
  const long tok = idx / ((long)HN * NG);
  const int  b_i = (int)(tok / SQ);
  const int  s   = (int)(tok % SQ);
  vT[(((long)b_i * NG + g) * HN + d) * SQ + s] =
      mixed[tok * QKV + QD + KD + (long)g * HN + d];
}

// ---------------- flash attention ----------------
// Block = 4 waves = 4 consecutive 16-row q tiles of ONE head. Waves share each
// 32-token K/V block, double-buffered in LDS via async global->LDS copies
// (ASYNCcnt), cutting K/V global traffic 4x. Causal ragged edge: late K blocks
// are skipped by early waves (barriers stay uniform).
__global__ __launch_bounds__(128)
void attn_k(const _Float16* __restrict__ q, const _Float16* __restrict__ k,
            const _Float16* __restrict__ vT, _Float16* __restrict__ ctx) {
  __shared__ alignas(64) _Float16 lds_k[2][32][128];  // 2 x 8KB, token-major
  __shared__ alignas(64) _Float16 lds_v[2][128][32];  // 2 x 8KB, hn-major
  __shared__ alignas(64) _Float16 lds_p[4][16][32];   // per-wave probs staging

  const int tid   = threadIdx.x;
  const int lane  = tid & 31;
  const int wslot = tid >> 5;
  const int nqb   = SQ / 64;                  // 32 q-blocks of 64 rows
  const int qb    = blockIdx.x % nqb;
  const int head  = (blockIdx.x / nqb) % NH;
  const int b_i   = blockIdx.x / (nqb * NH);
  const int g     = head / REP;
  const int qt    = qb * 4 + wslot;
  const int q0    = qt * 16;
  const int col   = lane & 15;
  const int hi    = lane >> 4;
  const int kb8   = hi * 8;
  const int kb16  = hi * 16;
  const float NEG_INF = -__builtin_inff();

  const _Float16* qh = q  + ((long)b_i * NH + head) * SQ * HN;
  const _Float16* kh = k  + ((long)b_i * NG + g)    * SQ * HN;
  const _Float16* vh = vT + ((long)b_i * NG + g)    * HN * SQ;

  // Q A-fragments for this wave's 16x128 tile (K-dim split 4x32)
  v16h aq[4];
  {
    const _Float16* rp = qh + (long)(q0 + col) * HN;
#pragma unroll
    for (int kk = 0; kk < 4; ++kk) {
      V16 u;
      u.h[0] = *(const v8h*)(rp + kk * 32 + kb8);
      u.h[1] = *(const v8h*)(rp + kk * 32 + 16 + kb8);
      aq[kk] = u.v;
    }
  }

  v8f acc[8];
#pragma unroll
  for (int t = 0; t < 8; ++t) acc[t] = (v8f){};
  float mrow[8], lrow[8];
#pragma unroll
  for (int r = 0; r < 8; ++r) { mrow[r] = NEG_INF; lrow[r] = 0.0f; }

  const int nkb_max = 2 * qb + 2;             // covers k tokens <= qb*64+63

  // stage K/V block kbs into LDS buffer (kbs & 1); block-wide cooperative copy
  auto stage = [&](int kbs) {
    _Float16* kl = &lds_k[kbs & 1][0][0];
    const _Float16* kg = kh + (long)kbs * 32 * HN;       // 8KB contiguous
    _Float16* vl = &lds_v[kbs & 1][tid][0];
    const _Float16* vg = vh + (long)tid * SQ + kbs * 32; // 64B per hn row
#pragma unroll
    for (int c = 0; c < 4; ++c) async_copy16(kg + tid * 32 + c * 8, kl + tid * 32 + c * 8);
#pragma unroll
    for (int c = 0; c < 4; ++c) async_copy16(vg + c * 8, vl + c * 8);
  };

  stage(0);
  for (int kb = 0; kb < nkb_max; ++kb) {
    const int kb0 = kb * 32;
    const int buf = kb & 1;
    async_drain();          // own async copies for block kb complete
    __syncthreads();        // everyone's copies complete; prior readers done
    if (kb + 1 < nkb_max) stage(kb + 1);   // prefetch next block into other buffer

    if (kb0 <= q0 + 15) {   // this wave still has unmasked k tokens here
      // scores: two 16x16 tiles over this 32-token block
      v8f s0 = (v8f){}, s1 = (v8f){};
#pragma unroll
      for (int kk = 0; kk < 4; ++kk) {
        V16 t0, t1;
        t0.h[0] = *(const v8h*)&lds_k[buf][col][kk * 32 + kb8];
        t0.h[1] = *(const v8h*)&lds_k[buf][col][kk * 32 + 16 + kb8];
        t1.h[0] = *(const v8h*)&lds_k[buf][16 + col][kk * 32 + kb8];
        t1.h[1] = *(const v8h*)&lds_k[buf][16 + col][kk * 32 + 16 + kb8];
        const v16h bk0 = t0.v;
        const v16h bk1 = t1.v;
        s0 = wmma_f16(aq[kk], bk0, s0);
        s1 = wmma_f16(aq[kk], bk1, s1);
      }

      const bool full = (kb0 + 31) <= q0;
#pragma unroll
      for (int r = 0; r < 8; ++r) {
        const int row = q0 + r + hi * 8;
        float v0 = s0[r], v1 = s1[r];
        if (!full) {
          if (kb0 + col      > row) v0 = NEG_INF;
          if (kb0 + 16 + col > row) v1 = NEG_INF;
        }
        float mx = fmaxf(v0, v1);
        mx = fmaxf(mx, __shfl_xor(mx, 1, 32));
        mx = fmaxf(mx, __shfl_xor(mx, 2, 32));
        mx = fmaxf(mx, __shfl_xor(mx, 4, 32));
        mx = fmaxf(mx, __shfl_xor(mx, 8, 32));
        const float mnew = fmaxf(mrow[r], mx);
        const float sc   = __expf(mrow[r] - mnew);
        const float p0   = __expf(v0 - mnew);
        const float p1   = __expf(v1 - mnew);
        float rs = p0 + p1;
        rs += __shfl_xor(rs, 1, 32);
        rs += __shfl_xor(rs, 2, 32);
        rs += __shfl_xor(rs, 4, 32);
        rs += __shfl_xor(rs, 8, 32);
        lrow[r] = lrow[r] * sc + rs;
        mrow[r] = mnew;
#pragma unroll
        for (int t = 0; t < 8; ++t) acc[t][r] *= sc;
        lds_p[wslot][r + hi * 8][col]      = (_Float16)p0;
        lds_p[wslot][r + hi * 8][col + 16] = (_Float16)p1;
      }

      // probs D-layout -> A-layout fixup through LDS (wave-private tile)
      V16 up;
      up.h[0] = *(const v8h*)(&lds_p[wslot][col][kb8]);
      up.h[1] = *(const v8h*)(&lds_p[wslot][col][16 + kb8]);
      const v16h pa = up.v;

      // ctx += P(16x32) * V(32x16) for 8 hn tiles; lds_v rows contiguous in tokens
#pragma unroll
      for (int t = 0; t < 8; ++t) {
        V16 tv;
        tv.h[0] = *(const v8h*)&lds_v[buf][t * 16 + col][kb8];
        tv.h[1] = *(const v8h*)&lds_v[buf][t * 16 + col][16 + kb8];
        const v16h bv = tv.v;
        acc[t] = wmma_f16(pa, bv, acc[t]);
      }
    }
  }

  // normalize and scatter ctx into [b, sq, NH*HN] f16
  _Float16* outp = ctx + (long)b_i * SQ * HH;
#pragma unroll
  for (int t = 0; t < 8; ++t) {
#pragma unroll
    for (int r = 0; r < 8; ++r) {
      const int m = q0 + r + hi * 8;
      const int c = head * HN + t * 16 + col;
      outp[(long)m * HH + c] = (_Float16)(acc[t][r] / lrow[r]);
    }
  }
}

// ---------------- host launch ----------------
extern "C" void kernel_launch(void* const* d_in, const int* in_sizes, int n_in,
                              void* d_out, int out_size, void* d_ws, size_t ws_size,
                              hipStream_t stream) {
  (void)in_sizes; (void)n_in; (void)out_size; (void)ws_size;
  const float* hidden  = (const float*)d_in[0];
  const float* cache   = (const float*)d_in[1];
  const float* w_qkv   = (const float*)d_in[2];
  const float* b_qkv   = (const float*)d_in[3];
  const float* w_dense = (const float*)d_in[4];

  char* ws = (char*)d_ws;
  _Float16* hid16   = (_Float16*)(ws);                      //  32 MB
  _Float16* wqkv16  = (_Float16*)(ws + 33554432UL);         //  36 MB
  _Float16* wd16    = (_Float16*)(ws + 71303168UL);         //  32 MB
  _Float16* mixed16 = (_Float16*)(ws + 104857600UL);        //  36 MB
  _Float16* q16     = (_Float16*)(ws + 142606336UL);        //  32 MB
  _Float16* k16     = (_Float16*)(ws + 176160768UL);        //   2 MB
  _Float16* vT16    = (_Float16*)(ws + 178257920UL);        //   2 MB
  _Float16* ctx16   = (_Float16*)(ws + 180355072UL);        //  32 MB

  cast_f32_f16_k<<<2048, 256, 0, stream>>>(hidden,  hid16,  (long)TOK * HH);
  cast_f32_f16_k<<<2048, 256, 0, stream>>>(w_qkv,   wqkv16, (long)QKV * HH);
  cast_f32_f16_k<<<2048, 256, 0, stream>>>(w_dense, wd16,   (long)HH * HH);

  // mixed = hid @ w_qkv^T + b_qkv   (f16 out)
  gemm_wmma_k<<<dim3(QKV / 256, TOK / 128), 256, 0, stream>>>(
      hid16, wqkv16, b_qkv, nullptr, mixed16, TOK, QKV, HH);

  const long nqk = (long)TOK * (NH + NG) * 64;
  rope_qk_k<<<(unsigned)((nqk + 255) / 256), 256, 0, stream>>>(mixed16, cache, q16, k16);

  const long nv = (long)TOK * NG * HN;
  v_trans_k<<<(unsigned)((nv + 255) / 256), 256, 0, stream>>>(mixed16, vT16);

  // one block per (b, head, 64 q rows); 4 waves share K/V blocks via LDS
  attn_k<<<BB * NH * (SQ / 64), 128, 0, stream>>>(q16, k16, vT16, ctx16);

  // out = ctx @ w_dense^T   (f32 out)
  gemm_wmma_k<<<dim3(HH / 256, TOK / 128), 256, 0, stream>>>(
      ctx16, wd16, nullptr, (float*)d_out, nullptr, TOK, HH, HH);
}
